// LSTM_73263552135272
// MI455X (gfx1250) — matrix operand order repaired and checked by reading, not compiled, not run on hardware
//
#include <hip/hip_runtime.h>
#include <hip/hip_bf16.h>

// LSTM (B=1024, T=512, IN=200, H=64) fused persistent kernel for gfx1250.
// One workgroup per 16-row batch tile for all 512 timesteps.
//  - Weights resident in VGPRs as WMMA B-fragments for the whole T loop.
//  - x tile staged (branch-free, 7 pairs/thread) to LDS bf16, double-buffered.
//  - All 9 A-fragments preloaded per step so ds_load latency overlaps the
//    18-deep v_wmma_f32_16x16x32_bf16 chain (partial s_wait_dscnt).
//  - fp32 accumulation, fp32 cell state/gate math in registers.

typedef __attribute__((ext_vector_type(16))) __bf16 v16bf;
typedef __attribute__((ext_vector_type(8)))  float  v8f;

#define T_STEPS 512
#define IN_DIM  200
#define HID     64
#define BT      16    // batch tile rows per workgroup

// LDS pitches padded to avoid bank conflicts (64 banks x 4B)
#define XPITCH  232   // bf16/row: 464B = 116 dwords, gcd(116,64)=4 -> clean
#define HPITCH  72    // bf16/row: 144B = 36 dwords,  gcd(36,64)=4  -> clean
#define GPITCH  260   // f32/row: offsets the two half-wave row groups

__device__ __forceinline__ float sigf(float x) {
    return 1.0f / (1.0f + __expf(-x));
}
__device__ __forceinline__ float tanhfast(float x) {
    return 2.0f / (1.0f + __expf(-2.0f * x)) - 1.0f;  // 2*sigmoid(2x)-1
}

// B-fragment (K x N = 32 x 16, bf16) of W^T from row-major W[rows][rowStride]:
// N=lane%16, element j<8 -> K=half*8+j, j>=8 -> K=16+half*8+(j-8) (matches A layout)
__device__ __forceinline__ v16bf load_bfrag(const float* __restrict__ W,
                                            int ncol, int rowStride, int k0,
                                            int kLimit, int half) {
    v16bf b;
#pragma unroll
    for (int j = 0; j < 16; ++j) {
        int k = k0 + ((j < 8) ? (half * 8 + j) : (16 + half * 8 + (j - 8)));
        float v = (k < kLimit) ? W[ncol * rowStride + k] : 0.0f;
        b[j] = (__bf16)v;
    }
    return b;
}

// A-fragment (16 x 32, bf16) from an LDS tile with row pitch `pitch`.
// ISA 7.12.2 16-bit A layout: M=lane%16; j<8 -> K=k0+half*8+j,
// j>=8 -> K=k0+16+half*8+(j-8). Each half is a 16B chunk -> ds_load_b128.
__device__ __forceinline__ v16bf load_afrag_lds(const __bf16* buf, int pitch,
                                                int m, int k0, int half) {
    v16bf a;
    const __bf16* p0 = buf + m * pitch + k0 + half * 8;
    const __bf16* p1 = p0 + 16;
#pragma unroll
    for (int j = 0; j < 8; ++j) {
        a[j]     = p0[j];
        a[j + 8] = p1[j];
    }
    return a;
}

__device__ __forceinline__ v8f wmma_bf16(v16bf a, v16bf b, v8f c) {
    return __builtin_amdgcn_wmma_f32_16x16x32_bf16(
        /*neg_a=*/false, a, /*neg_b=*/false, b,
        /*c_mod=*/(short)0, c, /*reuse_a=*/false, /*reuse_b=*/false);
}

__global__ __launch_bounds__(256, 1) void lstm_fused_kernel(
    const float* __restrict__ x,     // [B, T, IN]
    const float* __restrict__ W_ih,  // [4H, IN]
    const float* __restrict__ W_hh,  // [4H, H]
    const float* __restrict__ b_ih,  // [4H]
    const float* __restrict__ b_hh,  // [4H]
    const float* __restrict__ fc_w,  // [1, H]
    const float* __restrict__ fc_b,  // [1]
    float* __restrict__ out)         // [B, 1]
{
    __shared__ float  sGates[BT * GPITCH];     // ~16.6 KB gate exchange
    __shared__ __bf16 sHbf[BT * HPITCH];       //  ~2.3 KB h (WMMA A operand)
    __shared__ float  sHf[BT * HID];           //   4 KB h fp32 (final FC)
    __shared__ __bf16 sXbf[2][BT * XPITCH];    // ~14.5 KB double-buffered x tile

    const int tid  = threadIdx.x;
    const int wave = tid >> 5;          // 8 waves
    const int lane = tid & 31;          // wave32
    const int half = lane >> 4;
    const int lm   = lane & 15;
    const int b0   = blockIdx.x * BT;

    // Each wave owns gate columns [wave*32, wave*32+32)
    const int ncol0 = wave * 32 + lm;
    const int ncol1 = ncol0 + 16;

    // ---- Weights resident in VGPRs as B-fragments (loop-invariant) ----
    v16bf wih0[7], wih1[7];
#pragma unroll
    for (int kk = 0; kk < 7; ++kk) {
        wih0[kk] = load_bfrag(W_ih, ncol0, IN_DIM, kk * 32, IN_DIM, half);
        wih1[kk] = load_bfrag(W_ih, ncol1, IN_DIM, kk * 32, IN_DIM, half);
    }
    v16bf whh0[2], whh1[2];
#pragma unroll
    for (int kk = 0; kk < 2; ++kk) {
        whh0[kk] = load_bfrag(W_hh, ncol0, HID, kk * 32, HID, half);
        whh1[kk] = load_bfrag(W_hh, ncol1, HID, kk * 32, HID, half);
    }
    const float bias0 = b_ih[ncol0] + b_hh[ncol0];
    const float bias1 = b_ih[ncol1] + b_hh[ncol1];

    // ---- Branch-free staging setup: 7 pairs per thread, covers k in [0,224) ----
    const int srow = tid & 15;               // staged row
    const int sk   = 2 * (tid >> 4);         // pair start k: 0..30
    const float* sx_base = x + (size_t)(b0 + srow) * T_STEPS * IN_DIM;
    const int sk6  = sk + 192;               // tail pair k: 192..222
    const int sk6c = (sk6 <= 198) ? sk6 : 196;  // clamped (in-bounds) address
    const bool sk6v = (sk6 < 200);

    // ---- Init h = 0 in LDS, c = 0 in registers; stage tile 0 ----
    for (int i = tid; i < BT * HPITCH; i += 256) sHbf[i] = (__bf16)0.0f;
    for (int i = tid; i < BT * HID; i += 256)    sHf[i] = 0.0f;
    {
        const float* src = sx_base;          // t = 0
        __bf16* dst = &sXbf[0][srow * XPITCH + sk];
#pragma unroll
        for (int u = 0; u < 6; ++u) {
            float2 v = *reinterpret_cast<const float2*>(src + sk + 32 * u);
            dst[32 * u]     = (__bf16)v.x;
            dst[32 * u + 1] = (__bf16)v.y;
        }
        float2 v6 = *reinterpret_cast<const float2*>(src + sk6c);
        dst[192] = (__bf16)(sk6v ? v6.x : 0.0f);
        dst[193] = (__bf16)(sk6v ? v6.y : 0.0f);
    }

    const int urow = tid >> 4;          // batch row for the update phase
    const int ucol = (tid & 15) * 4;    // 4 consecutive hidden cells
    float creg[4] = {0.0f, 0.0f, 0.0f, 0.0f};
    __syncthreads();

    // ---- Sequential recurrence over T ----
    for (int t = 0; t < T_STEPS; ++t) {
        const __bf16* xt = sXbf[t & 1];

        // 1) Preload ALL A-fragments: 18 back-to-back ds_load_b128 so the
        //    WMMA chain below can overlap LDS latency via partial dscnt waits.
        v16bf ax[7];
#pragma unroll
        for (int kk = 0; kk < 7; ++kk)
            ax[kk] = load_afrag_lds(xt, XPITCH, lm, kk * 32, half);
        v16bf ah[2];
#pragma unroll
        for (int kk = 0; kk < 2; ++kk)
            ah[kk] = load_afrag_lds(sHbf, HPITCH, lm, kk * 32, half);

        // 2) Stage tile t+1 (global loads issue early; its ds_stores follow
        //    the fragment ds_loads in DS order) + prefetch tile t+2 to L2.
        if (t + 1 < T_STEPS) {
            const float* src = sx_base + (size_t)(t + 1) * IN_DIM;
            __bf16* dst = &sXbf[(t + 1) & 1][srow * XPITCH + sk];
#pragma unroll
            for (int u = 0; u < 6; ++u) {
                float2 v = *reinterpret_cast<const float2*>(src + sk + 32 * u);
                dst[32 * u]     = (__bf16)v.x;
                dst[32 * u + 1] = (__bf16)v.y;
            }
            float2 v6 = *reinterpret_cast<const float2*>(src + sk6c);
            dst[192] = (__bf16)(sk6v ? v6.x : 0.0f);
            dst[193] = (__bf16)(sk6v ? v6.y : 0.0f);
        }
        if (t + 2 < T_STEPS) {
            const int pk = (tid >> 4) * 13;  // 0..195: one addr per cacheline
            __builtin_prefetch(sx_base + (size_t)(t + 2) * IN_DIM + pk, 0, 3);
        }

        // 3) gates = x_t @ W_ih^T (7 steps) + h @ W_hh^T (2 steps)
        v8f acc0 = {0.f, 0.f, 0.f, 0.f, 0.f, 0.f, 0.f, 0.f};
        v8f acc1 = {0.f, 0.f, 0.f, 0.f, 0.f, 0.f, 0.f, 0.f};
#pragma unroll
        for (int kk = 0; kk < 7; ++kk) {
            acc0 = wmma_bf16(ax[kk], wih0[kk], acc0);
            acc1 = wmma_bf16(ax[kk], wih1[kk], acc1);
        }
#pragma unroll
        for (int kk = 0; kk < 2; ++kk) {
            acc0 = wmma_bf16(ah[kk], whh0[kk], acc0);
            acc1 = wmma_bf16(ah[kk], whh1[kk], acc1);
        }

        // 4) Spill gate pre-activations. C/D layout: elem r -> M=half*8+r.
#pragma unroll
        for (int r = 0; r < 8; ++r) {
            sGates[(half * 8 + r) * GPITCH + ncol0] = acc0[r] + bias0;
            sGates[(half * 8 + r) * GPITCH + ncol1] = acc1[r] + bias1;
        }
        __syncthreads();  // gates + staged tile visible; fences h reads/writes

        // 5) Elementwise LSTM cell update (fp32): 4 cells per thread.
#pragma unroll
        for (int q = 0; q < 4; ++q) {
            const int col = ucol + q;
            const float ig = sGates[urow * GPITCH + 0   + col];
            const float fg = sGates[urow * GPITCH + 64  + col];
            const float gg = sGates[urow * GPITCH + 128 + col];
            const float og = sGates[urow * GPITCH + 192 + col];
            float c = sigf(fg) * creg[q] + sigf(ig) * tanhfast(gg);
            float h = sigf(og) * tanhfast(c);
            creg[q] = c;
            sHbf[urow * HPITCH + col] = (__bf16)h;
            sHf[urow * HID + col]     = h;
        }
        __syncthreads();  // new h visible for next step
    }

    // ---- Final FC + sigmoid: out[b] = sigmoid(h_T[b,:] . fc_w + fc_b) ----
    if (tid < BT) {
        float acc = fc_b[0];
#pragma unroll 4
        for (int k = 0; k < HID; ++k)
            acc += sHf[tid * HID + k] * fc_w[k];
        out[b0 + tid] = sigf(acc);
    }
}

extern "C" void kernel_launch(void* const* d_in, const int* in_sizes, int n_in,
                              void* d_out, int out_size, void* d_ws, size_t ws_size,
                              hipStream_t stream) {
    const float* x    = (const float*)d_in[0];
    const float* W_ih = (const float*)d_in[1];
    const float* W_hh = (const float*)d_in[2];
    const float* b_ih = (const float*)d_in[3];
    const float* b_hh = (const float*)d_in[4];
    const float* fc_w = (const float*)d_in[5];
    const float* fc_b = (const float*)d_in[6];
    float* out = (float*)d_out;

    const int B = 1024;
    dim3 grid(B / BT);   // 64 workgroups
    dim3 block(256);     // 8 wave32 waves
    lstm_fused_kernel<<<grid, block, 0, stream>>>(x, W_ih, W_hh, b_ih, b_hh,
                                                  fc_w, fc_b, out);
}
